// MovieExpertCRS_48318382080147
// MI455X (gfx1250) — matrix operand order repaired and checked by reading, not compiled, not run on hardware
//
#include <hip/hip_runtime.h>
#include <hip/hip_bf16.h>

typedef __attribute__((ext_vector_type(16))) _Float16 v16h;
typedef __attribute__((ext_vector_type(8)))  _Float16 v8h;
typedef __attribute__((ext_vector_type(8)))  float    v8f;

#define NE    30000
#define NR    12
#define NB    8
#define KD    128
#define TD    768
#define BATCH 32
#define LCE   50
#define SEQL  256
#define NEDGE 400000

// ---------------- helpers ----------------
__device__ __forceinline__ float wave_half16_sum(float v) {
  // sum within each group of 16 lanes (wave32)
  #pragma unroll
  for (int off = 1; off < 16; off <<= 1) v += __shfl_xor(v, off, 32);
  return v;
}

template <int NW>
__device__ __forceinline__ float blk_sum(float v, float* red, int tid) {
  #pragma unroll
  for (int off = 16; off; off >>= 1) v += __shfl_xor(v, off, 32);
  if ((tid & 31) == 0) red[tid >> 5] = v;
  __syncthreads();
  float s = 0.f;
  #pragma unroll
  for (int i = 0; i < NW; ++i) s += red[i];
  __syncthreads();
  return s;
}

template <int NW>
__device__ __forceinline__ float blk_max(float v, float* red, int tid) {
  #pragma unroll
  for (int off = 16; off; off >>= 1) v = fmaxf(v, __shfl_xor(v, off, 32));
  if ((tid & 31) == 0) red[tid >> 5] = v;
  __syncthreads();
  float s = -1e30f;
  #pragma unroll
  for (int i = 0; i < NW; ++i) s = fmaxf(s, red[i]);
  __syncthreads();
  return s;
}

// ---------------- K1: zero scratch ----------------
__global__ void zero_kernel(float* p, long n) {
  long i = (long)blockIdx.x * blockDim.x + threadIdx.x;
  long stride = (long)gridDim.x * blockDim.x;
  for (; i < n; i += stride) p[i] = 0.f;
}

// ---------------- K2: per-(rel,dst) edge counts ----------------
__global__ void count_kernel(const int* __restrict__ ei, const int* __restrict__ et,
                             int* __restrict__ cnt) {
  int i = blockIdx.x * blockDim.x + threadIdx.x;
  int stride = gridDim.x * blockDim.x;
  for (; i < NEDGE; i += stride) {
    int r = et[i];
    int dst = ei[NEDGE + i];
    atomicAdd(&cnt[r * NE + dst], 1);
  }
}

// ---------------- K3: fused message + mean-scatter ----------------
// msg[e,:] = (1/cnt) * sum_b comp[r,b] * basis[b, src, :]  -> atomicAdd agg[dst,:]
__global__ void rgcn_scatter_kernel(const int* __restrict__ ei, const int* __restrict__ et,
                                    const float* __restrict__ comp,
                                    const float* __restrict__ basis,
                                    const int* __restrict__ cnt,
                                    float* __restrict__ agg) {
  int e = blockIdx.x * 4 + (threadIdx.x >> 7);   // 4 edges per 512-thread block
  int d = threadIdx.x & 127;
  if (e >= NEDGE) return;
  int r = et[e];
  int src = ei[e];
  int dst = ei[NEDGE + e];
  float norm = 1.f / fmaxf((float)cnt[r * NE + dst], 1.f);
  float m = 0.f;
  #pragma unroll
  for (int b = 0; b < NB; ++b)
    m += comp[r * NB + b] * basis[((long)b * NE + src) * KD + d];
  atomicAdd(&agg[(long)dst * KD + d], m * norm);
}

// ---------------- K4: kg = agg + root + bias ; kg_h = f16(kg) ----------------
__global__ void kg_finalize_kernel(float* __restrict__ agg, const float* __restrict__ root,
                                   const float* __restrict__ bias, _Float16* __restrict__ kg_h) {
  long n = (long)NE * KD;
  long i = (long)blockIdx.x * blockDim.x + threadIdx.x;
  long stride = (long)gridDim.x * blockDim.x;
  for (; i < n; i += stride) {
    float v = agg[i] + root[i] + bias[i & (KD - 1)];
    agg[i] = v;
    kg_h[i] = (_Float16)v;
  }
}

// ---------------- K5: tok_W -> f16, transposed (Wt[n][k] = W[k][n]) ----------------
__global__ void convW_kernel(const float* __restrict__ W, _Float16* __restrict__ Wt) {
  long n = (long)TD * TD;
  long i = (long)blockIdx.x * blockDim.x + threadIdx.x;
  long stride = (long)gridDim.x * blockDim.x;
  for (; i < n; i += stride) {
    long nn = i / TD, k = i % TD;
    Wt[i] = (_Float16)W[k * TD + nn];
  }
}

// ---------------- K6: entity self-dot attention pool (per batch) ----------------
__global__ void ent_attn_kernel(const int* __restrict__ ctx, const float* __restrict__ kg,
                                const float* __restrict__ W, const float* __restrict__ bvec,
                                const float* __restrict__ vvec, float* __restrict__ ent_rep) {
  __shared__ float reps[LCE][KD];
  __shared__ float evals[LCE];
  __shared__ float red[4];
  int b = blockIdx.x, tid = threadIdx.x;
  for (int idx = tid; idx < LCE * KD; idx += 128) {
    int l = idx >> 7, d = idx & 127;
    int ent = ctx[b * LCE + l];
    reps[l][d] = kg[(long)ent * KD + d];
  }
  __syncthreads();
  for (int l = 0; l < LCE; ++l) {
    float acc = bvec[tid];
    #pragma unroll 4
    for (int d = 0; d < KD; ++d) acc += reps[l][d] * W[d * KD + tid];
    float t = tanhf(acc) * vvec[tid];
    float s = blk_sum<4>(t, red, tid);
    if (tid == 0) evals[l] = s;
  }
  __syncthreads();
  // masked softmax over L=50
  bool valid = (tid < LCE) && (ctx[b * LCE + tid] != 0);
  float ev = (tid < LCE) ? (valid ? evals[tid] : -1e9f) : -1e30f;
  float mx = blk_max<4>(ev, red, tid);
  float ex = (tid < LCE) ? __expf(ev - mx) : 0.f;
  float sm = blk_sum<4>(ex, red, tid);
  if (tid < LCE) evals[tid] = ex / sm;
  __syncthreads();
  float out = 0.f;
  for (int l = 0; l < LCE; ++l) out += evals[l] * reps[l][tid];
  ent_rep[b * KD + tid] = out;
}

// ---------------- K7: WMMA token-attention logits ----------------
// e[row] += sum_n v[n] * tanh( (h @ W)[row,n] + b[n] ),  M=8192, N=K=768
// Block = 256 threads (8 waves), M_TILE = 64 rows staged in LDS.
// Waves arranged 4 (M-strips) x 2 (N-tile halves): four waves walk the same
// N-tile at the same time -> 4-way L0 reuse of the B (Wt) stream.
#define MT 64
__global__ void tok_e_wmma_kernel(const float* __restrict__ h, const _Float16* __restrict__ Wt,
                                  const float* __restrict__ bvec, const float* __restrict__ vvec,
                                  float* __restrict__ e_tok) {
  extern __shared__ __align__(16) _Float16 Ash[];   // [MT][TD+8]
  const int LDW = TD + 8;
  int tid = threadIdx.x;
  int m0 = blockIdx.x * MT;
  // stage MT x 768 A strip (f32 -> f16) in LDS
  for (int idx = tid; idx < MT * TD; idx += 256) {
    int r = idx / TD, c = idx - r * TD;
    Ash[r * LDW + c] = (_Float16)h[(long)(m0 + r) * TD + c];
  }
  __syncthreads();
  int lane = tid & 31, wave = tid >> 5;
  int m = lane & 15, g = lane >> 4;
  int mstrip = wave & 3;            // which 16-row strip of the 64
  int ntbase = (wave >> 2) * 6;     // ntiles [0..5] or [6..11]
  const _Float16* Arow = Ash + (mstrip * 16 + m) * LDW;
  for (int t = 0; t < 6; ++t) {
    int n = (ntbase + t) * 16 + m;
    const _Float16* wrow = Wt + (long)n * TD;
    v8f acc = {};
    #pragma unroll 2
    for (int k0 = 0; k0 < TD; k0 += 32) {
      if ((k0 & 63) == 0) __builtin_prefetch(wrow + k0 + 256, 0, 1);
      v8h alo = *(const v8h*)(Arow + k0 + 8 * g);
      v8h ahi = *(const v8h*)(Arow + k0 + 16 + 8 * g);
      v8h blo = *(const v8h*)(wrow + k0 + 8 * g);
      v8h bhi = *(const v8h*)(wrow + k0 + 16 + 8 * g);
      v16h a  = __builtin_shufflevector(alo, ahi, 0,1,2,3,4,5,6,7,8,9,10,11,12,13,14,15);
      v16h bm = __builtin_shufflevector(blo, bhi, 0,1,2,3,4,5,6,7,8,9,10,11,12,13,14,15);
      acc = __builtin_amdgcn_wmma_f32_16x16x32_f16(false, a, false, bm, (short)0, acc,
                                                   false, false);
    }
    float bn = bvec[n], vn = vvec[n];
    int rowbase = m0 + mstrip * 16;
    #pragma unroll
    for (int r = 0; r < 8; ++r) {
      float val = tanhf(acc[r] + bn) * vn;   // element (M = rowbase+r+8g, N = n)
      val = wave_half16_sum(val);            // sum over the 16 N in this tile
      if (m == 0) atomicAdd(&e_tok[rowbase + r + 8 * g], val);
    }
  }
}

// ---------------- K8: token masked softmax + weighted pool (per batch) ----------------
__global__ void tok_pool_kernel(const int* __restrict__ toks, const float* __restrict__ h,
                                const float* __restrict__ e_tok, float* __restrict__ tok_rep) {
  __shared__ float a_sh[SEQL];
  __shared__ float red[8];
  int b = blockIdx.x, tid = threadIdx.x;
  bool valid = toks[b * SEQL + tid] != 0;
  float ev = valid ? e_tok[b * SEQL + tid] : -1e9f;
  float mx = blk_max<8>(ev, red, tid);
  float ex = __expf(ev - mx);
  float sm = blk_sum<8>(ex, red, tid);
  a_sh[tid] = ex / sm;
  __syncthreads();
  for (int d = tid; d < TD; d += SEQL) {
    float acc = 0.f;
    #pragma unroll 4
    for (int l = 0; l < SEQL; ++l) acc += a_sh[l] * h[((long)b * SEQL + l) * TD + d];
    tok_rep[b * TD + d] = acc;
  }
}

// ---------------- K9: lin + gate fusion -> user_h (f16) ----------------
__global__ void fuse_kernel(const float* __restrict__ tok_rep, const float* __restrict__ ent_rep,
                            const float* __restrict__ linW, const float* __restrict__ linb,
                            const float* __restrict__ gateW, const float* __restrict__ gateb,
                            _Float16* __restrict__ user_h) {
  __shared__ float catv[2 * KD];
  int b = blockIdx.x, d = threadIdx.x;
  float acc = linb[d];
  const float* tr = tok_rep + (long)b * TD;
  #pragma unroll 4
  for (int k = 0; k < TD; ++k) acc += tr[k] * linW[(long)d * TD + k];
  catv[d] = acc;                     // token part first (matches concat order)
  catv[KD + d] = ent_rep[b * KD + d];
  __syncthreads();
  float gv = gateb[d];
  #pragma unroll 4
  for (int j = 0; j < 2 * KD; ++j) gv += gateW[d * 2 * KD + j] * catv[j];
  gv = 1.f / (1.f + __expf(-gv));
  float u = gv * catv[d] + (1.f - gv) * catv[KD + d];
  user_h[b * KD + d] = (_Float16)u;
}

// ---------------- K10: WMMA scores = user @ kg^T  (M=32, K=128, N=30000) ----------------
__global__ void scores_wmma_kernel(const _Float16* __restrict__ user_h,
                                   const _Float16* __restrict__ kg_h,
                                   float* __restrict__ out) {
  int lane = threadIdx.x & 31, wave = threadIdx.x >> 5;
  int ntile = blockIdx.x * 8 + wave;          // 1875 tiles of 16 entities
  if (ntile >= NE / 16) return;               // wave-uniform branch
  int m = lane & 15, g = lane >> 4;
  int n = ntile * 16 + m;
  const _Float16* krow = kg_h + (long)n * KD;
  v8f acc0 = {}, acc1 = {};
  #pragma unroll 2
  for (int k0 = 0; k0 < KD; k0 += 32) {
    v8h blo = *(const v8h*)(krow + k0 + 8 * g);
    v8h bhi = *(const v8h*)(krow + k0 + 16 + 8 * g);
    v16h bm = __builtin_shufflevector(blo, bhi, 0,1,2,3,4,5,6,7,8,9,10,11,12,13,14,15);
    v8h a0lo = *(const v8h*)(user_h + (long)m * KD + k0 + 8 * g);
    v8h a0hi = *(const v8h*)(user_h + (long)m * KD + k0 + 16 + 8 * g);
    v8h a1lo = *(const v8h*)(user_h + (long)(m + 16) * KD + k0 + 8 * g);
    v8h a1hi = *(const v8h*)(user_h + (long)(m + 16) * KD + k0 + 16 + 8 * g);
    v16h a0 = __builtin_shufflevector(a0lo, a0hi, 0,1,2,3,4,5,6,7,8,9,10,11,12,13,14,15);
    v16h a1 = __builtin_shufflevector(a1lo, a1hi, 0,1,2,3,4,5,6,7,8,9,10,11,12,13,14,15);
    acc0 = __builtin_amdgcn_wmma_f32_16x16x32_f16(false, a0, false, bm, (short)0, acc0,
                                                  false, false);
    acc1 = __builtin_amdgcn_wmma_f32_16x16x32_f16(false, a1, false, bm, (short)0, acc1,
                                                  false, false);
  }
  #pragma unroll
  for (int r = 0; r < 8; ++r) {
    out[(long)(r + 8 * g) * NE + n] = acc0[r];
    out[(long)(16 + r + 8 * g) * NE + n] = acc1[r];
  }
}

// ---------------- launch ----------------
extern "C" void kernel_launch(void* const* d_in, const int* in_sizes, int n_in,
                              void* d_out, int out_size, void* d_ws, size_t ws_size,
                              hipStream_t stream) {
  const int*   ctx_ent = (const int*)d_in[0];
  const int*   ctx_tok = (const int*)d_in[1];
  const int*   ei      = (const int*)d_in[2];
  const int*   et      = (const int*)d_in[3];
  const float* tok_emb = (const float*)d_in[4];
  const float* comp    = (const float*)d_in[5];
  const float* basis   = (const float*)d_in[6];
  const float* root    = (const float*)d_in[7];
  const float* rbias   = (const float*)d_in[8];
  const float* entW    = (const float*)d_in[9];
  const float* entb    = (const float*)d_in[10];
  const float* entv    = (const float*)d_in[11];
  const float* tokW    = (const float*)d_in[12];
  const float* tokb    = (const float*)d_in[13];
  const float* tokv    = (const float*)d_in[14];
  const float* linW    = (const float*)d_in[15];
  const float* linb    = (const float*)d_in[16];
  const float* gateW   = (const float*)d_in[17];
  const float* gateb   = (const float*)d_in[18];
  float* scores = (float*)d_out;

  // workspace carve-up (all 256B aligned)
  char* base = (char*)d_ws;
  size_t off = 0;
  auto carve = [&](size_t bytes) {
    char* p = base + off;
    off = (off + bytes + 255) & ~(size_t)255;
    return p;
  };
  int*      cnt     = (int*)     carve((size_t)NR * NE * 4);        // 1.44 MB
  float*    agg     = (float*)   carve((size_t)NE * KD * 4);        // 15.36 MB (becomes kg)
  _Float16* kg_h    = (_Float16*)carve((size_t)NE * KD * 2);        // 7.68 MB
  _Float16* Wt_h    = (_Float16*)carve((size_t)TD * TD * 2);        // 1.18 MB
  float*    e_tok   = (float*)   carve((size_t)BATCH * SEQL * 4);
  float*    ent_rep = (float*)   carve((size_t)BATCH * KD * 4);
  float*    tok_rep = (float*)   carve((size_t)BATCH * TD * 4);
  _Float16* user_h  = (_Float16*)carve((size_t)BATCH * KD * 2);
  (void)ws_size; (void)n_in; (void)in_sizes; (void)out_size;

  // 1) zero accumulators (cnt zeroed via float-bitpattern 0)
  zero_kernel<<<2048, 256, 0, stream>>>((float*)cnt, (long)NR * NE);
  zero_kernel<<<4096, 256, 0, stream>>>(agg, (long)NE * KD);
  zero_kernel<<<32, 256, 0, stream>>>(e_tok, (long)BATCH * SEQL);
  // 2) edge counts
  count_kernel<<<1024, 256, 0, stream>>>(ei, et, cnt);
  // 3) RGCN message + mean scatter (basis stays L2-resident: 123 MB < 192 MB L2)
  rgcn_scatter_kernel<<<(NEDGE + 3) / 4, 512, 0, stream>>>(ei, et, comp, basis, cnt, agg);
  // 4) kg = agg + root + bias, f16 copy for WMMA B-fragments
  kg_finalize_kernel<<<4096, 256, 0, stream>>>(agg, root, rbias, kg_h);
  // 5) tok_W -> f16 transposed (B-fragment friendly)
  convW_kernel<<<2048, 256, 0, stream>>>(tokW, Wt_h);
  // 6) entity attention pool
  ent_attn_kernel<<<BATCH, 128, 0, stream>>>(ctx_ent, agg, entW, entb, entv, ent_rep);
  // 7) WMMA token attention logits (fused GEMM + tanh + v-dot reduction)
  size_t lds_bytes = (size_t)MT * (TD + 8) * sizeof(_Float16);   // ~97 KB dynamic LDS
  tok_e_wmma_kernel<<<(BATCH * SEQL) / MT, 256, lds_bytes, stream>>>(tok_emb, Wt_h, tokb,
                                                                     tokv, e_tok);
  // 8) token masked softmax + pooled sum
  tok_pool_kernel<<<BATCH, SEQL, 0, stream>>>(ctx_tok, tok_emb, e_tok, tok_rep);
  // 9) linear + gate fusion -> user embedding (f16 A-fragments)
  fuse_kernel<<<BATCH, KD, 0, stream>>>(tok_rep, ent_rep, linW, linb, gateW, gateb, user_h);
  // 10) WMMA scores = user @ kg^T
  scores_wmma_kernel<<<(NE / 16 + 7) / 8, 256, 0, stream>>>(user_h, kg_h, scores);
}
